// DehazeBlock_5677946765566
// MI455X (gfx1250) — matrix operand order repaired and checked by reading, not compiled, not run on hardware
//
#include <hip/hip_runtime.h>
#include <hip/hip_bf16.h>

typedef __attribute__((ext_vector_type(16))) __bf16 v16bf;
typedef __attribute__((ext_vector_type(8)))  float  v8f;

#define HWPIX 65536
#define IMG   256

// ---------------------------------------------------------------------------
// f32 -> bf16 conversion
// ---------------------------------------------------------------------------
__global__ __launch_bounds__(256)
void cvt_bf16(const float* __restrict__ in, __bf16* __restrict__ out, int n) {
    int i = blockIdx.x * 256 + threadIdx.x;
    if (i < n) out[i] = (__bf16)in[i];
}

// ---------------------------------------------------------------------------
// Pack conv weights [O=64,I=64,3,3] into per-lane WMMA A-fragment order.
// Fragment index space: tap t(0..8) x kstep ks(0..1) x ogroup og(0..3)
//                       x lane L(0..31) x elem j(0..15)
// 16-bit A 16x32 layout used here: M = L%16, half = L/16,
//   K = half*16 + j (contiguous, matches B fragment loads)
// ---------------------------------------------------------------------------
__global__ __launch_bounds__(256)
void pack_conv_w(const float* __restrict__ w, __bf16* __restrict__ out) {
    int idx = blockIdx.x * 256 + threadIdx.x;
    if (idx >= 9 * 2 * 4 * 32 * 16) return;
    int j  = idx & 15;
    int L  = (idx >> 4) & 31;
    int og = (idx >> 9) & 3;
    int ks = (idx >> 11) & 1;
    int t  = idx >> 12;
    int half = L >> 4, m = L & 15;
    int K = half * 16 + j;
    int c = ks * 32 + K;
    int o = og * 16 + m;
    int ky = t / 3, kx = t - ky * 3;
    out[idx] = (__bf16)w[((o * 64 + c) * 3 + ky) * 3 + kx];
}

// Pack an 8x64 matrix (cv_w / pa_w1) as a 16x64 A operand (rows 8..15 zero).
__global__ __launch_bounds__(256)
void pack_a8(const float* __restrict__ w, __bf16* __restrict__ out) {
    int idx = blockIdx.x * 256 + threadIdx.x;
    if (idx >= 2 * 32 * 16) return;
    int j  = idx & 15;
    int L  = (idx >> 4) & 31;
    int ks = idx >> 9;
    int half = L >> 4, m = L & 15;
    int K = half * 16 + j;
    int c = ks * 32 + K;
    out[idx] = (m < 8) ? (__bf16)w[m * 64 + c] : (__bf16)0.f;
}

// ---------------------------------------------------------------------------
// Per-(b,c) channel mean over H*W of a bf16 NCHW tensor -> f32 [B*C]
// ---------------------------------------------------------------------------
__global__ __launch_bounds__(256)
void chan_mean(const __bf16* __restrict__ in, float* __restrict__ out) {
    __shared__ float red[256];
    const int bc = blockIdx.x, tid = threadIdx.x;
    const __bf16* p = in + (size_t)bc * HWPIX;
    float s = 0.f;
    for (int i = tid; i < HWPIX; i += 256) s += (float)p[i];
    red[tid] = s; __syncthreads();
    for (int k = 128; k > 0; k >>= 1) { if (tid < k) red[tid] += red[tid + k]; __syncthreads(); }
    if (tid == 0) out[bc] = red[0] * (1.f / (float)HWPIX);
}

// ---------------------------------------------------------------------------
// conv3x3 + bias (+relu) via implicit GEMM (9 shifted 1x1 GEMMs), bf16 WMMA.
// Block: 256 thr (8 waves) -> one (batch, 16x16 spatial tile), all 64 out ch.
// LDS: 18x18x64 bf16 halo tile, [y][x][c] so B-fragments are 32B contiguous.
// ---------------------------------------------------------------------------
__global__ __launch_bounds__(256)
void conv3x3_wmma(const __bf16* __restrict__ in, const v16bf* __restrict__ wp,
                  const float* __restrict__ bias, __bf16* __restrict__ out,
                  int do_relu) {
    __shared__ __bf16 tile[18 * 18 * 64];
    const int tx0 = blockIdx.x * 16, ty0 = blockIdx.y * 16, b = blockIdx.z;
    const int tid = threadIdx.x;

    for (int idx = tid; idx < 64 * 18 * 18; idx += 256) {
        int c = idx / 324;
        int r = idx - c * 324;
        int y = r / 18, x = r - y * 18;
        int gy = ty0 - 1 + y, gx = tx0 - 1 + x;
        float v = 0.f;
        if ((unsigned)gy < (unsigned)IMG && (unsigned)gx < (unsigned)IMG)
            v = (float)in[(size_t)(b * 64 + c) * HWPIX + gy * IMG + gx];
        tile[(y * 18 + x) * 64 + c] = (__bf16)v;
    }
    __syncthreads();

    const int wid = tid >> 5, lane = tid & 31;
    const int n = lane & 15, half = lane >> 4;
    for (int tIdx = wid; tIdx < 64; tIdx += 8) {
        const int og = tIdx >> 4;   // out-channel group (16 channels)
        const int g  = tIdx & 15;   // output row within tile, lane n = column
        v8f acc = {};
        for (int t = 0; t < 9; ++t) {
            const int ky = t / 3, kx = t - ky * 3;
            for (int ks = 0; ks < 2; ++ks) {
                v16bf a = wp[(((t * 2 + ks) * 4 + og) * 32) + lane];
                v16bf bm = *(const v16bf*)&tile[((g + ky) * 18 + (n + kx)) * 64
                                                + ks * 32 + half * 16];
                acc = __builtin_amdgcn_wmma_f32_16x16x32_bf16(
                    false, a, false, bm, (short)0, acc, false, false);
            }
        }
        for (int r = 0; r < 8; ++r) {
            int o = og * 16 + r + 8 * half;             // D layout: M = r + 8*half
            float v = acc[r] + bias[o];
            if (do_relu) v = fmaxf(v, 0.f);
            out[(size_t)(b * 64 + o) * HWPIX + (ty0 + g) * IMG + tx0 + n] = (__bf16)v;
        }
    }
}

// ---------------------------------------------------------------------------
// TDP scalar preamble (single block, 64 threads): two_weight + fc/fc0/fc1 +
// collapse of c1/c2 GEMMs into A1/A2 (per b,o) and rowsums s1/s2.
// S layout (floats): [0..1]=mix w1,w2  [2..257]=A1  [258..513]=A2
//                    [514..577]=s1     [578..641]=s2
// ---------------------------------------------------------------------------
__global__ __launch_bounds__(64)
void tdp_prep(const float* __restrict__ Mres, const float* __restrict__ Mx,
              const float* __restrict__ twf_w, const float* __restrict__ twf_b,
              const float* __restrict__ fc_w,  const float* __restrict__ fc_b,
              const float* __restrict__ fc0_w, const float* __restrict__ fc0_b,
              const float* __restrict__ fc1_w, const float* __restrict__ fc1_b,
              const float* __restrict__ c1_w,  const float* __restrict__ c1_b,
              const float* __restrict__ c2_w,  const float* __restrict__ c2_b,
              float* __restrict__ S) {
    __shared__ float tws[2], temp[4][64], t1s[4][8], a0s[4][64], a1s[4][64];
    const int tid = threadIdx.x;
    if (tid == 0) {
        float m1 = 0.f, m2 = 0.f;
        for (int c = 0; c < 64; ++c) { m1 += Mres[c]; m2 += Mx[c]; }
        m1 *= (1.f / 64.f); m2 *= (1.f / 64.f);      // batch-0 means (ref takes y[0])
        float w0 = fmaxf(twf_w[0] * m1 + twf_w[1] * m2 + twf_b[0], 0.f);
        float w1 = fmaxf(twf_w[2] * m1 + twf_w[3] * m2 + twf_b[1], 0.f);
        tws[0] = w0; tws[1] = w1; S[0] = w0; S[1] = w1;
    }
    __syncthreads();
    for (int b = 0; b < 4; ++b)
        temp[b][tid] = tws[0] * Mres[b * 64 + tid] + tws[1] * Mx[b * 64 + tid];
    __syncthreads();
    if (tid < 8)
        for (int b = 0; b < 4; ++b) {
            float s = fc_b[tid];
            for (int c = 0; c < 64; ++c) s += fc_w[tid * 64 + c] * temp[b][c];
            t1s[b][tid] = fmaxf(s, 0.f);
        }
    __syncthreads();
    for (int b = 0; b < 4; ++b) {
        float s0 = fc0_b[tid], s1 = fc1_b[tid];
        for (int r = 0; r < 8; ++r) {
            s0 += fc0_w[tid * 8 + r] * t1s[b][r];
            s1 += fc1_w[tid * 8 + r] * t1s[b][r];
        }
        a0s[b][tid] = s0; a1s[b][tid] = s1;
    }
    __syncthreads();
    float rs1 = 0.f, rs2 = 0.f;
    for (int c = 0; c < 64; ++c) { rs1 += c1_w[tid * 64 + c]; rs2 += c2_w[tid * 64 + c]; }
    S[514 + tid] = rs1; S[578 + tid] = rs2;
    for (int b = 0; b < 4; ++b) {
        float s1 = c1_b[tid], s2 = c2_b[tid];
        for (int c = 0; c < 64; ++c) {
            s1 += c1_w[tid * 64 + c] * a0s[b][c];
            s2 += c2_w[tid * 64 + c] * a1s[b][c];
        }
        S[2 + b * 64 + tid] = s1; S[258 + b * 64 + tid] = s2;
    }
}

// CA preamble: ca[b,o] = sigmoid(ca_w2 @ relu(ca_w1 @ mean + b1) + b2)
__global__ __launch_bounds__(64)
void ca_prep(const float* __restrict__ M,
             const float* __restrict__ w1, const float* __restrict__ b1,
             const float* __restrict__ w2, const float* __restrict__ b2,
             float* __restrict__ ca) {
    __shared__ float t[4][8];
    const int tid = threadIdx.x;
    if (tid < 8)
        for (int b = 0; b < 4; ++b) {
            float s = b1[tid];
            for (int c = 0; c < 64; ++c) s += w1[tid * 64 + c] * M[b * 64 + c];
            t[b][tid] = fmaxf(s, 0.f);
        }
    __syncthreads();
    for (int b = 0; b < 4; ++b) {
        float s = b2[tid];
        for (int r = 0; r < 8; ++r) s += w2[tid * 8 + r] * t[b][r];
        ca[b * 64 + tid] = 1.f / (1.f + __expf(-s));
    }
}

// ---------------------------------------------------------------------------
// Fused TDP apply: xm = w1*x1 + w2*x2 ; t2 = relu(cv @ xm) (WMMA, 8 rows
// padded to 16) ; b0/b1 = cv0/cv1 dots ; softmax-of-2 combine.
// Block: 128 thr (4 wave32) -> 64 pixels of one batch.
// ---------------------------------------------------------------------------
template <typename OutT>
__global__ __launch_bounds__(128)
void tdp_apply(const __bf16* __restrict__ x1, const __bf16* __restrict__ x2,
               const float* __restrict__ S, const v16bf* __restrict__ ap,
               const float* __restrict__ cv_b,
               const float* __restrict__ cv0_w, const float* __restrict__ cv0_b,
               const float* __restrict__ cv1_w, const float* __restrict__ cv1_b,
               OutT* __restrict__ out) {
    __shared__ __bf16 xm[64 * 64];
    __shared__ float bb[64 * 2];
    const int b  = blockIdx.x >> 10;
    const int p0 = (blockIdx.x & 1023) * 64;
    const int tid = threadIdx.x;
    const float w1 = S[0], w2 = S[1];

    for (int i = tid; i < 4096; i += 128) {
        int pix = i & 63, c = i >> 6;
        size_t idx = (size_t)(b * 64 + c) * HWPIX + p0 + pix;
        xm[pix * 64 + c] = (__bf16)(w1 * (float)x1[idx] + w2 * (float)x2[idx]);
    }
    __syncthreads();
    {
        const int wid = tid >> 5, lane = tid & 31;
        const int n = lane & 15, half = lane >> 4;
        const int pix = wid * 16 + n;
        v8f acc = {};
        for (int ks = 0; ks < 2; ++ks) {
            v16bf a  = ap[ks * 32 + lane];
            v16bf bm = *(const v16bf*)&xm[pix * 64 + ks * 32 + half * 16];
            acc = __builtin_amdgcn_wmma_f32_16x16x32_bf16(
                false, a, false, bm, (short)0, acc, false, false);
        }
        if (half == 0) {                 // valid t2 rows 0..7 live in half=0 lanes
            float b0 = cv0_b[0], b1 = cv1_b[0];
            for (int r = 0; r < 8; ++r) {
                float t = fmaxf(acc[r] + cv_b[r], 0.f);
                b0 += cv0_w[r] * t; b1 += cv1_w[r] * t;
            }
            bb[pix * 2] = b0; bb[pix * 2 + 1] = b1;
        }
    }
    __syncthreads();
    for (int i = tid; i < 4096; i += 128) {
        int pix = i & 63, c = i >> 6;
        size_t idx = (size_t)(b * 64 + c) * HWPIX + p0 + pix;
        float z = S[2 + b * 64 + c] - S[258 + b * 64 + c]
                + bb[pix * 2] * S[514 + c] - bb[pix * 2 + 1] * S[578 + c];
        float wa = 1.f / (1.f + __expf(-z));        // softmax over 2 == sigmoid(y1-y2)
        float o  = wa * (float)x1[idx] + (1.f - wa) * (float)x2[idx];
        out[idx] = (OutT)o;
    }
}

// ---------------------------------------------------------------------------
// Fused CA * PA: xm = res2*ca[b,c]; t = relu(pa_w1 @ xm) (WMMA);
// y = sigmoid(pa_w2 . t + pa_b2); out = xm * y
// ---------------------------------------------------------------------------
__global__ __launch_bounds__(128)
void ca_pa(const __bf16* __restrict__ in, const float* __restrict__ ca,
           const v16bf* __restrict__ ap, const float* __restrict__ pa_b1,
           const float* __restrict__ pa_w2, const float* __restrict__ pa_b2,
           __bf16* __restrict__ out) {
    __shared__ __bf16 xm[64 * 64];
    __shared__ float yy[64];
    const int b  = blockIdx.x >> 10;
    const int p0 = (blockIdx.x & 1023) * 64;
    const int tid = threadIdx.x;

    for (int i = tid; i < 4096; i += 128) {
        int pix = i & 63, c = i >> 6;
        size_t idx = (size_t)(b * 64 + c) * HWPIX + p0 + pix;
        xm[pix * 64 + c] = (__bf16)((float)in[idx] * ca[b * 64 + c]);
    }
    __syncthreads();
    {
        const int wid = tid >> 5, lane = tid & 31;
        const int n = lane & 15, half = lane >> 4;
        const int pix = wid * 16 + n;
        v8f acc = {};
        for (int ks = 0; ks < 2; ++ks) {
            v16bf a  = ap[ks * 32 + lane];
            v16bf bm = *(const v16bf*)&xm[pix * 64 + ks * 32 + half * 16];
            acc = __builtin_amdgcn_wmma_f32_16x16x32_bf16(
                false, a, false, bm, (short)0, acc, false, false);
        }
        if (half == 0) {
            float s = pa_b2[0];
            for (int r = 0; r < 8; ++r) s += pa_w2[r] * fmaxf(acc[r] + pa_b1[r], 0.f);
            yy[pix] = 1.f / (1.f + __expf(-s));
        }
    }
    __syncthreads();
    for (int i = tid; i < 4096; i += 128) {
        int pix = i & 63, c = i >> 6;
        size_t idx = (size_t)(b * 64 + c) * HWPIX + p0 + pix;
        out[idx] = (__bf16)((float)xm[pix * 64 + c] * yy[pix]);
    }
}

// ---------------------------------------------------------------------------
extern "C" void kernel_launch(void* const* d_in, const int* in_sizes, int n_in,
                              void* d_out, int out_size, void* d_ws, size_t ws_size,
                              hipStream_t stream) {
    auto F = [&](int i) { return (const float*)d_in[i]; };
    const float* x       = F(0);
    const float* conv1_w = F(1);  const float* conv1_b = F(2);
    const float* conv2_w = F(3);  const float* conv2_b = F(4);
    const float* ca_w1 = F(5);  const float* ca_b1 = F(6);
    const float* ca_w2 = F(7);  const float* ca_b2 = F(8);
    const float* pa_w1 = F(9);  const float* pa_b1 = F(10);
    const float* pa_w2 = F(11); const float* pa_b2 = F(12);
    const float* twf1_w = F(13); const float* twf1_b = F(14);
    const float* twf2_w = F(15); const float* twf2_b = F(16);
    const int T1 = 17, T2 = 33;  // tdp param bases
    // within tdp: 0 fc_w 1 fc_b 2 cv_w 3 cv_b 4 fc0_w 5 fc0_b 6 fc1_w 7 fc1_b
    //             8 cv0_w 9 cv0_b 10 cv1_w 11 cv1_b 12 c1_w 13 c1_b 14 c2_w 15 c2_b

    char* ws = (char*)d_ws;
    const size_t TEN = (size_t)4 * 64 * HWPIX * sizeof(__bf16);   // 33,554,432
    __bf16* xbf  = (__bf16*)(ws);
    __bf16* bufA = (__bf16*)(ws + TEN);
    __bf16* bufB = (__bf16*)(ws + 2 * TEN);
    size_t o = 3 * TEN;
    __bf16* wp1  = (__bf16*)(ws + o); o += 131072;
    __bf16* wp2  = (__bf16*)(ws + o); o += 131072;
    __bf16* ap1  = (__bf16*)(ws + o); o += 4096;
    __bf16* ap2  = (__bf16*)(ws + o); o += 4096;
    __bf16* appa = (__bf16*)(ws + o); o += 4096;
    float*  Mx   = (float*)(ws + o);  o += 4096;
    float*  Ma   = (float*)(ws + o);  o += 4096;
    float*  Mb   = (float*)(ws + o);  o += 4096;
    float*  S1f  = (float*)(ws + o);  o += 4096;
    float*  S2f  = (float*)(ws + o);  o += 4096;
    float*  CAf  = (float*)(ws + o);  o += 4096;

    const int NPIX = 4 * 64 * HWPIX;

    cvt_bf16<<<(NPIX + 255) / 256, 256, 0, stream>>>(x, xbf, NPIX);
    pack_conv_w<<<144, 256, 0, stream>>>(conv1_w, wp1);
    pack_conv_w<<<144, 256, 0, stream>>>(conv2_w, wp2);
    pack_a8<<<4, 256, 0, stream>>>(F(T1 + 2), ap1);   // tdp1 cv_w
    pack_a8<<<4, 256, 0, stream>>>(F(T2 + 2), ap2);   // tdp2 cv_w
    pack_a8<<<4, 256, 0, stream>>>(pa_w1, appa);
    chan_mean<<<256, 256, 0, stream>>>(xbf, Mx);

    // res = relu(conv1(x))
    conv3x3_wmma<<<dim3(16, 16, 4), 256, 0, stream>>>(xbf, (const v16bf*)wp1, conv1_b, bufA, 1);
    chan_mean<<<256, 256, 0, stream>>>(bufA, Ma);
    tdp_prep<<<1, 64, 0, stream>>>(Ma, Mx, twf1_w, twf1_b,
        F(T1 + 0), F(T1 + 1), F(T1 + 4), F(T1 + 5), F(T1 + 6), F(T1 + 7),
        F(T1 + 12), F(T1 + 13), F(T1 + 14), F(T1 + 15), S1f);
    // res1 = tdp_attention(res, x)
    tdp_apply<__bf16><<<4096, 128, 0, stream>>>(bufA, xbf, S1f, (const v16bf*)ap1,
        F(T1 + 3), F(T1 + 8), F(T1 + 9), F(T1 + 10), F(T1 + 11), bufB);

    // res2 = conv2(res1)
    conv3x3_wmma<<<dim3(16, 16, 4), 256, 0, stream>>>(bufB, (const v16bf*)wp2, conv2_b, bufA, 0);
    chan_mean<<<256, 256, 0, stream>>>(bufA, Ma);
    ca_prep<<<1, 64, 0, stream>>>(Ma, ca_w1, ca_b1, ca_w2, ca_b2, CAf);
    // res3 = PA(CA(res2))
    ca_pa<<<4096, 128, 0, stream>>>(bufA, CAf, (const v16bf*)appa, pa_b1, pa_w2, pa_b2, bufB);

    chan_mean<<<256, 256, 0, stream>>>(bufB, Mb);
    tdp_prep<<<1, 64, 0, stream>>>(Mb, Mx, twf2_w, twf2_b,
        F(T2 + 0), F(T2 + 1), F(T2 + 4), F(T2 + 5), F(T2 + 6), F(T2 + 7),
        F(T2 + 12), F(T2 + 13), F(T2 + 14), F(T2 + 15), S2f);
    // res4 = tdp_attention(res3, x) -> f32 output
    tdp_apply<float><<<4096, 128, 0, stream>>>(bufB, xbf, S2f, (const v16bf*)ap2,
        F(T2 + 3), F(T2 + 8), F(T2 + 9), F(T2 + 10), F(T2 + 11), (float*)d_out);
}
